// pointHead_44616120271066
// MI455X (gfx1250) — compile-verified, matrix operand order checked by reading, not run on hardware
//
#include <hip/hip_runtime.h>
#include <hip/hip_fp16.h>

typedef __attribute__((ext_vector_type(16))) _Float16 v16h;
typedef __attribute__((ext_vector_type(8)))  float    v8f;

#define B_    8
#define POVER 768
#define NPTS  256
#define BETA  192
#define NCOV  64
#define C1    128
#define H1    256
#define W1    256
#define C2    64
#define H2    512
#define W2    512
#define KC    192   // 128 coarse + 64 fine channels

// ---------------------------------------------------------------------------
// Kernel 1: uncertainty at oversampled points.
// Per point: top-2 over 128 channels at each of the 4 bilinear corner pixels,
// then bilinear blend (zero padding for invalid corners), u = -(top1 - top2).
// ---------------------------------------------------------------------------
__global__ void k_uncertainty(const float* __restrict__ outf,
                              const float* __restrict__ rand_over,
                              float* __restrict__ u) {
    int idx = blockIdx.x * blockDim.x + threadIdx.x;
    if (idx >= B_ * POVER) return;
    int b = idx / POVER;
    float px = rand_over[(size_t)idx * 2 + 0];
    float py = rand_over[(size_t)idx * 2 + 1];
    float gx = px * W1 - 0.5f, gy = py * H1 - 0.5f;
    float x0f = floorf(gx), y0f = floorf(gy);
    float wx = gx - x0f, wy = gy - y0f;
    int x0 = (int)x0f, y0 = (int)y0f;
    const float* base = outf + (size_t)b * C1 * H1 * W1;
    const float wgt[4] = {(1.f-wx)*(1.f-wy), wx*(1.f-wy), (1.f-wx)*wy, wx*wy};
    float og1 = 0.f, og2 = 0.f;
    for (int cr = 0; cr < 4; ++cr) {
        int xi = x0 + (cr & 1);
        int yi = y0 + (cr >> 1);
        if (xi < 0 || xi >= W1 || yi < 0 || yi >= H1) continue;  // zero pad
        const float* pix = base + (size_t)yi * W1 + xi;
        float m1 = -INFINITY, m2 = -INFINITY;
        for (int c = 0; c < C1; ++c) {
            float v = pix[(size_t)c * (H1 * W1)];
            if (v > m1) { m2 = m1; m1 = v; }
            else if (v > m2) { m2 = v; }
        }
        og1 += wgt[cr] * m1;
        og2 += wgt[cr] * m2;
    }
    u[idx] = -(og1 - og2);
}

// ---------------------------------------------------------------------------
// Kernel 2: stable descending top-k (k=BETA) by rank counting in LDS.
// rank(i) = #{j : u[j] > u[i]} + #{j < i : u[j] == u[i]}  (matches lax.top_k)
// Writes selected rand_over points + rand_cov tail directly into `points`.
// One block per batch.
// ---------------------------------------------------------------------------
__global__ void k_select(const float* __restrict__ u,
                         const float* __restrict__ rand_over,
                         const float* __restrict__ rand_cov,
                         float* __restrict__ points) {
    __shared__ float su[POVER];
    int b = blockIdx.x;
    int t = threadIdx.x;  // 256 threads
    for (int i = t; i < POVER; i += 256) su[i] = u[(size_t)b * POVER + i];
    __syncthreads();
    for (int i = t; i < POVER; i += 256) {
        float ui = su[i];
        int rank = 0;
        for (int j = 0; j < POVER; ++j) {
            float uj = su[j];
            rank += (uj > ui) || (uj == ui && j < i);
        }
        if (rank < BETA) {
            points[((size_t)b * NPTS + rank) * 2 + 0] = rand_over[((size_t)b * POVER + i) * 2 + 0];
            points[((size_t)b * NPTS + rank) * 2 + 1] = rand_over[((size_t)b * POVER + i) * 2 + 1];
        }
    }
    if (t < NCOV) {
        points[((size_t)b * NPTS + BETA + t) * 2 + 0] = rand_cov[((size_t)b * NCOV + t) * 2 + 0];
        points[((size_t)b * NPTS + BETA + t) * 2 + 1] = rand_cov[((size_t)b * NCOV + t) * 2 + 1];
    }
}

// ---------------------------------------------------------------------------
// Bilinear sample of a single [H,W] plane (align_corners=False, zero pad).
// ---------------------------------------------------------------------------
__device__ __forceinline__ float bilin(const float* __restrict__ plane,
                                       int H, int W, float px, float py) {
    float gx = px * W - 0.5f, gy = py * H - 0.5f;
    float x0f = floorf(gx), y0f = floorf(gy);
    float wx = gx - x0f, wy = gy - y0f;
    int x0 = (int)x0f, y0 = (int)y0f;
    float acc = 0.f;
#pragma unroll
    for (int cr = 0; cr < 4; ++cr) {
        int xi = x0 + (cr & 1);
        int yi = y0 + (cr >> 1);
        float w = ((cr & 1) ? wx : 1.f - wx) * ((cr >> 1) ? wy : 1.f - wy);
        if (xi >= 0 && xi < W && yi >= 0 && yi < H)
            acc += w * plane[(size_t)yi * W + xi];
    }
    return acc;
}

// ---------------------------------------------------------------------------
// Kernel 3: gather featT[b][n][c] (c<128: out, else res2), convert to f16.
// K-contiguous layout so GEMM B-fragment loads become b128; c-fastest thread
// mapping so the f16 writes are fully coalesced and all threads of one point
// broadcast the same points[] read.
// ---------------------------------------------------------------------------
__global__ void k_feat(const float* __restrict__ outf,
                       const float* __restrict__ res2,
                       const float* __restrict__ points,
                       _Float16* __restrict__ featT) {
    int idx = blockIdx.x * blockDim.x + threadIdx.x;  // [b][n][c] flat
    if (idx >= B_ * NPTS * KC) return;
    int c = idx % KC;
    int n = (idx / KC) % NPTS;
    int b = idx / (KC * NPTS);
    float px = points[((size_t)b * NPTS + n) * 2 + 0];
    float py = points[((size_t)b * NPTS + n) * 2 + 1];
    float v;
    if (c < C1) v = bilin(outf + ((size_t)b * C1 + c) * (H1 * W1), H1, W1, px, py);
    else        v = bilin(res2 + ((size_t)b * C2 + (c - C1)) * (H2 * W2), H2, W2, px, py);
    featT[idx] = (_Float16)v;
}

// ---------------------------------------------------------------------------
// Kernel 4: convert weight [128,192] to f16.
// ---------------------------------------------------------------------------
__global__ void k_wcvt(const float* __restrict__ w, _Float16* __restrict__ wf) {
    int i = blockIdx.x * blockDim.x + threadIdx.x;
    if (i < 128 * KC) wf[i] = (_Float16)w[i];
}

// ---------------------------------------------------------------------------
// Kernel 5: WMMA GEMM  rend[b] = W(128x192) * featT[b]^T(192x256) + bias.
// One wave per 16x16 output tile; K=192 = 6 x v_wmma_f32_16x16x32_f16.
// A 16x32 f16 layout (ISA 7.12.2): lanes 0-15 & 16-31 both hold M=lane&15;
// VGPR v<4 -> K = v*2 + half*8, v>=4 -> K = 16 + (v-4)*2 + half*8 (half=lane>>4).
// B 32x16 symmetric (N = lane&15, same K mapping). C/D: VGPR r -> M = r+half*8.
// Both A rows (K-contiguous in wf) and B columns (K-contiguous in featT) merge
// into global_load_b128; 384B row stride keeps 16B alignment.
// ---------------------------------------------------------------------------
__global__ void k_gemm(const _Float16* __restrict__ wf,
                       const _Float16* __restrict__ featT,
                       const float* __restrict__ bias,
                       float* __restrict__ rend) {
    int wave = (blockIdx.x * blockDim.x + threadIdx.x) >> 5;  // 0..1023
    int lane = threadIdx.x & 31;
    int b  = wave >> 7;         // 8 batches
    int tm = (wave >> 4) & 7;   // 8 M-tiles
    int tn = wave & 15;         // 16 N-tiles
    int half = lane >> 4;
    int mn   = lane & 15;

    const _Float16* wrow = wf    + (size_t)(tm * 16 + mn) * KC;              // A row M
    const _Float16* fcol = featT + ((size_t)b * NPTS + tn * 16 + mn) * KC;   // B col N

    v8f acc = {};
#pragma unroll
    for (int kk = 0; kk < KC; kk += 32) {
        v16h a, bm;
#pragma unroll
        for (int v = 0; v < 8; ++v) {
            int kb = kk + ((v < 4) ? v * 2 : 16 + (v - 4) * 2) + half * 8;
            a[2 * v]      = wrow[kb];
            a[2 * v + 1]  = wrow[kb + 1];
            bm[2 * v]     = fcol[kb];
            bm[2 * v + 1] = fcol[kb + 1];
        }
        acc = __builtin_amdgcn_wmma_f32_16x16x32_f16(
            /*neg_a=*/false, a, /*neg_b=*/false, bm,
            /*c_mod=*/(short)0, acc, /*reuse_a=*/false, /*reuse_b=*/false);
    }
#pragma unroll
    for (int r = 0; r < 8; ++r) {
        int M = tm * 16 + r + half * 8;
        int N = tn * 16 + mn;
        rend[((size_t)b * C1 + M) * NPTS + N] = acc[r] + bias[M];
    }
}

// ---------------------------------------------------------------------------
extern "C" void kernel_launch(void* const* d_in, const int* in_sizes, int n_in,
                              void* d_out, int out_size, void* d_ws, size_t ws_size,
                              hipStream_t stream) {
    // setup_inputs order: x, res2, out, rand_over, rand_cov, weight, bias
    const float* res2   = (const float*)d_in[1];
    const float* outf   = (const float*)d_in[2];
    const float* rover  = (const float*)d_in[3];
    const float* rcov   = (const float*)d_in[4];
    const float* weight = (const float*)d_in[5];
    const float* bias   = (const float*)d_in[6];

    float* rend   = (float*)d_out;                       // [8,128,256]
    float* points = rend + (size_t)B_ * C1 * NPTS;       // [8,256,2] appended

    char* ws = (char*)d_ws;
    float*    u     = (float*)ws;                          //  24 KB
    _Float16* wf16  = (_Float16*)(ws + 32768);             //  48 KB
    _Float16* featT = (_Float16*)(ws + 32768 + 65536);     // 768 KB  [b][n][c]

    k_uncertainty<<<(B_ * POVER + 127) / 128, 128, 0, stream>>>(outf, rover, u);
    k_wcvt<<<(128 * KC + 255) / 256, 256, 0, stream>>>(weight, wf16);
    k_select<<<B_, 256, 0, stream>>>(u, rover, rcov, points);
    k_feat<<<(B_ * NPTS * KC + 255) / 256, 256, 0, stream>>>(outf, res2, points, featT);
    k_gemm<<<(1024 * 32) / 256, 256, 0, stream>>>(wf16, featT, bias, rend);
}
